// BinaryDenseLayer_20220706030267
// MI455X (gfx1250) — compile-verified, matrix operand order checked by reading, not run on hardware
//
#include <hip/hip_runtime.h>

typedef _Float16 hf;
typedef __attribute__((ext_vector_type(16))) _Float16 v16h;
typedef __attribute__((ext_vector_type(8)))  _Float16 v8h;
typedef __attribute__((ext_vector_type(4)))  _Float16 v4h;
typedef __attribute__((ext_vector_type(8)))  float    v8f;

#define TM 128
#define TN 128
#define TK 32
#define LDSK 40   // TK + 8 pad: 80-byte row stride, keeps 16B alignment, staggers banks

// C[M,N] = X[M,K] * sign(W[K,N]) + b[N], f16 WMMA with f32 accumulate.
__global__ __launch_bounds__(256) void binary_dense_wmma_f16(
    const float* __restrict__ X,
    const float* __restrict__ W,
    const float* __restrict__ bias,
    float* __restrict__ C,
    int M, int N, int K)
{
    __shared__ hf sA[TM * LDSK];          // [row][k]  row-major f16
    __shared__ hf sB[TN * LDSK];          // [col][k]  transposed, binarized W

    const int tid  = threadIdx.x;
    const int lane = tid & 31;
    const int wave = tid >> 5;            // 0..7 (8 waves, wave32)
    const int wm   = wave >> 2;           // 0..1 : 64 rows of C each
    const int wn   = wave & 3;            // 0..3 : 32 cols of C each
    const int hi   = lane >> 4;           // lane half (K-interleave select)
    const int l16  = lane & 15;

    const int rowBase = blockIdx.y * TM;
    const int colBase = blockIdx.x * TN;

    v8f acc[4][2] = {};                   // 4 m-tiles x 2 n-tiles of 16x16 f32

    for (int k0 = 0; k0 < K; k0 += TK) {
        // ---- Stage A tile: 128x32 f32 -> f16 into sA (row-major) ----
#pragma unroll
        for (int i = 0; i < 4; ++i) {
            int idx = tid + i * 256;                  // 0..1023 float4 slots
            int r   = idx >> 3;                       // 0..127
            int c4  = idx & 7;                        // 0..7
            const float4 v = *(const float4*)(X + (size_t)(rowBase + r) * K + k0 + c4 * 4);
            v4h h;
            h[0] = (hf)v.x; h[1] = (hf)v.y; h[2] = (hf)v.z; h[3] = (hf)v.w;
            *(v4h*)(&sA[r * LDSK + c4 * 4]) = h;
        }
        // ---- Stage B tile: 32x128 f32 W -> sign() f16, transposed into sB[col][k] ----
#pragma unroll
        for (int i = 0; i < 4; ++i) {
            int idx = tid + i * 256;                  // 0..1023 float4 slots
            int kr  = idx >> 5;                       // 0..31  (K row)
            int c4  = idx & 31;                       // 0..31  (col group of 4)
            const float4 v = *(const float4*)(W + (size_t)(k0 + kr) * N + colBase + c4 * 4);
            sB[(c4 * 4 + 0) * LDSK + kr] = (hf)((v.x > 0.f) ? 1.f : ((v.x < 0.f) ? -1.f : 0.f));
            sB[(c4 * 4 + 1) * LDSK + kr] = (hf)((v.y > 0.f) ? 1.f : ((v.y < 0.f) ? -1.f : 0.f));
            sB[(c4 * 4 + 2) * LDSK + kr] = (hf)((v.z > 0.f) ? 1.f : ((v.z < 0.f) ? -1.f : 0.f));
            sB[(c4 * 4 + 3) * LDSK + kr] = (hf)((v.w > 0.f) ? 1.f : ((v.w < 0.f) ? -1.f : 0.f));
        }
        __syncthreads();

        // ---- Fragments per ISA wave32 16-bit A/B layout:
        // lanes 0-15: K 0..7 in V0-3, K 16..23 in V4-7; lanes 16-31: K 8..15, 24..31.
        v16h afrag[4], bfrag[2];
#pragma unroll
        for (int mt = 0; mt < 4; ++mt) {
            const hf* p = &sA[(wm * 64 + mt * 16 + l16) * LDSK + hi * 8];
            union { v16h v; v8h h[2]; } u;
            u.h[0] = *(const v8h*)p;          // K base + {0..7}
            u.h[1] = *(const v8h*)(p + 16);   // K base + 16 + {0..7}
            afrag[mt] = u.v;
        }
#pragma unroll
        for (int nt = 0; nt < 2; ++nt) {
            const hf* p = &sB[(wn * 32 + nt * 16 + l16) * LDSK + hi * 8];
            union { v16h v; v8h h[2]; } u;
            u.h[0] = *(const v8h*)p;
            u.h[1] = *(const v8h*)(p + 16);
            bfrag[nt] = u.v;
        }

#pragma unroll
        for (int mt = 0; mt < 4; ++mt)
#pragma unroll
            for (int nt = 0; nt < 2; ++nt)
                acc[mt][nt] = __builtin_amdgcn_wmma_f32_16x16x32_f16(
                    false, afrag[mt], false, bfrag[nt],
                    (short)0, acc[mt][nt], false, false);

        __syncthreads();
    }

    // ---- Epilogue: D layout (lanes 0-15: M=j, N=lane; lanes 16-31: M=j+8, N=lane-16) ----
#pragma unroll
    for (int mt = 0; mt < 4; ++mt) {
#pragma unroll
        for (int nt = 0; nt < 2; ++nt) {
            const int col = colBase + wn * 32 + nt * 16 + l16;
            const float bv = bias[col];
#pragma unroll
            for (int j = 0; j < 8; ++j) {
                const int row = rowBase + wm * 64 + mt * 16 + hi * 8 + j;
                C[(size_t)row * N + col] = acc[mt][nt][j] + bv;
            }
        }
    }
}

extern "C" void kernel_launch(void* const* d_in, const int* in_sizes, int n_in,
                              void* d_out, int out_size, void* d_ws, size_t ws_size,
                              hipStream_t stream) {
    const float* X = (const float*)d_in[0];   // [M, K] f32
    const float* W = (const float*)d_in[1];   // [K, N] f32
    const float* b = (const float*)d_in[2];   // [N]    f32
    float* C = (float*)d_out;                 // [M, N] f32

    const int N = in_sizes[2];
    const int K = in_sizes[1] / N;
    const int M = in_sizes[0] / K;

    dim3 grid(N / TN, M / TM);                // 32 x 32 blocks for 4096^2
    binary_dense_wmma_f16<<<grid, 256, 0, stream>>>(X, W, b, C, M, N, K);
}
